// GatedEquivariantBlock_73907797229833
// MI455X (gfx1250) — matrix operand order repaired and checked by reading, not compiled, not run on hardware
//
#include <hip/hip_runtime.h>
#include <math.h>

// Problem constants (from the reference): N=20000, E=320000, NS=128, NV=64, NT=64.
#define N_NODES 20000
#define N_EDGES 320000

typedef __attribute__((ext_vector_type(16))) __bf16 v16bf;
typedef __attribute__((ext_vector_type(8)))  float  v8f;

__device__ __forceinline__ float silu_f(float x)    { return x / (1.0f + __expf(-x)); }
__device__ __forceinline__ float sigmoid_f(float x) { return 1.0f / (1.0f + __expf(-x)); }

__device__ __forceinline__ v8f wmma_bf16(v16bf a, v16bf b, v8f c) {
  // emits v_wmma_f32_16x16x32_bf16
  return __builtin_amdgcn_wmma_f32_16x16x32_bf16(false, a, false, b, (short)0, c, false, false);
}

// A fragment: 16x32 tile (rows m0..m0+15, cols k0..k0+31) of a row-major f32
// matrix, converted to bf16 in registers.
// This code's A k order (same as B): lane L holds row m = L&15; lane-half selects K chunk:
//   slots 0..15 -> K = k0 + half*16 + s
__device__ __forceinline__ v16bf load_a_frag_f32(const float* __restrict__ src, int ld,
                                                 int m0, int k0, int lane) {
  const int m = m0 + (lane & 15);
  const int half = lane >> 4;
  const float* p = src + (size_t)m * ld + k0 + half * 16;
  float4 x0 = *(const float4*)(p + 0);
  float4 x1 = *(const float4*)(p + 4);
  float4 x2 = *(const float4*)(p + 8);
  float4 x3 = *(const float4*)(p + 12);
  v16bf a;
  a[0]  = (__bf16)x0.x; a[1]  = (__bf16)x0.y; a[2]  = (__bf16)x0.z; a[3]  = (__bf16)x0.w;
  a[4]  = (__bf16)x1.x; a[5]  = (__bf16)x1.y; a[6]  = (__bf16)x1.z; a[7]  = (__bf16)x1.w;
  a[8]  = (__bf16)x2.x; a[9]  = (__bf16)x2.y; a[10] = (__bf16)x2.z; a[11] = (__bf16)x2.w;
  a[12] = (__bf16)x3.x; a[13] = (__bf16)x3.y; a[14] = (__bf16)x3.z; a[15] = (__bf16)x3.w;
  return a;
}

// B fragment: 32x16 tile of W^T where W is [Nout,K] row-major bf16.
// B[k,n] = W[n,k]; lane L holds column n = n0 + (L&15), K chunk 16*(L>>4):
// 16 consecutive bf16 of one weight row = one 32-byte load.
__device__ __forceinline__ v16bf load_b_frag(const __bf16* __restrict__ W, int ld,
                                             int n0, int k0, int lane) {
  const int n = n0 + (lane & 15);
  const int koff = k0 + ((lane >> 4) << 4);
  return *(const v16bf*)(W + (size_t)n * ld + koff);
}

// Prefetch the next tile's contiguous 16-row A block (lane-strided 128B).
__device__ __forceinline__ void prefetch_tile(const float* __restrict__ X, int ld,
                                              int tile, int ntiles, int lane) {
  if (tile < ntiles) {
    const char* p = (const char*)(X + (size_t)tile * 16 * ld);
    __builtin_prefetch(p + lane * 128, 0, 1);                       // 4KB
    if (ld >= 128) __builtin_prefetch(p + 4096 + lane * 128, 0, 1); // next 4KB
    if (ld >= 192) __builtin_prefetch(p + 8192 + lane * 128, 0, 1); // next 4KB
  }
}

// ---------------------------------------------------------------------------
// hs = silu(h_ns @ Wns1^T + b1) @ Wns2^T + b2  -> plain store (initializes acc)
// Persistent waves; W2 fragments + biases hoisted out of the tile loop.
// ---------------------------------------------------------------------------
__global__ __launch_bounds__(256) void node_scalar_kernel(
    const float* __restrict__ X,                 // [M,128]
    const __bf16* __restrict__ W1, const float* __restrict__ b1,  // [64,128],[64]
    const __bf16* __restrict__ W2, const float* __restrict__ b2,  // [64,64],[64]
    float* __restrict__ acc_out, int ntiles)     // [M,64]
{
  __shared__ float lds[8][16][68];
  const int wave = threadIdx.x >> 5, lane = threadIdx.x & 31;
  const int wtid = blockIdx.x * 8 + wave;
  const int wstride = gridDim.x * 8;
  const int half = lane >> 4, ln = lane & 15;
  float (*t)[68] = lds[wave];

  v16bf B2[2][4];
  float bias1[4], bias2[4];
#pragma unroll
  for (int ks = 0; ks < 2; ++ks)
#pragma unroll
    for (int nt = 0; nt < 4; ++nt)
      B2[ks][nt] = load_b_frag(W2, 64, nt * 16, ks * 32, lane);
#pragma unroll
  for (int nt = 0; nt < 4; ++nt) { bias1[nt] = b1[nt * 16 + ln]; bias2[nt] = b2[nt * 16 + ln]; }

  for (int tile = wtid; tile < ntiles; tile += wstride) {
    const int m0 = tile * 16;
    prefetch_tile(X, 128, tile + wstride, ntiles, lane);
    __builtin_amdgcn_wave_barrier();
    v8f acc[4] = {};
#pragma unroll
    for (int ks = 0; ks < 4; ++ks) {
      v16bf a = load_a_frag_f32(X, 128, m0, ks * 32, lane);
#pragma unroll
      for (int nt = 0; nt < 4; ++nt)
        acc[nt] = wmma_bf16(a, load_b_frag(W1, 128, nt * 16, ks * 32, lane), acc[nt]);
    }
#pragma unroll
    for (int nt = 0; nt < 4; ++nt)
#pragma unroll
      for (int r = 0; r < 8; ++r)
        t[r + 8 * half][nt * 16 + ln] = silu_f(acc[nt][r] + bias1[nt]);
    __builtin_amdgcn_wave_barrier();
    v8f acc2[4] = {};
#pragma unroll
    for (int ks = 0; ks < 2; ++ks) {
      v16bf a = load_a_frag_f32(&t[0][0], 68, 0, ks * 32, lane);
#pragma unroll
      for (int nt = 0; nt < 4; ++nt)
        acc2[nt] = wmma_bf16(a, B2[ks][nt], acc2[nt]);
    }
#pragma unroll
    for (int nt = 0; nt < 4; ++nt)
#pragma unroll
      for (int r = 0; r < 8; ++r)
        acc_out[(size_t)(m0 + r + 8 * half) * 64 + nt * 16 + ln] = acc2[nt][r] + bias2[nt];
  }
}

// ---------------------------------------------------------------------------
// es = silu(h_es @ Wes1^T + b1) @ Wes2^T + b2, fused scatter-add to both nodes
// Persistent waves; W2 fragments + biases hoisted.
// ---------------------------------------------------------------------------
__global__ __launch_bounds__(256) void edge_scalar_kernel(
    const float* __restrict__ X,                 // [E,128]
    const int* __restrict__ idx1, const int* __restrict__ idx2,
    const __bf16* __restrict__ W1, const float* __restrict__ b1,
    const __bf16* __restrict__ W2, const float* __restrict__ b2,
    float* __restrict__ node_acc, int ntiles)    // [N,64]
{
  __shared__ float lds[8][16][68];
  const int wave = threadIdx.x >> 5, lane = threadIdx.x & 31;
  const int wtid = blockIdx.x * 8 + wave;
  const int wstride = gridDim.x * 8;
  const int half = lane >> 4, ln = lane & 15;
  float (*t)[68] = lds[wave];

  v16bf B2[2][4];
  float bias1[4], bias2[4];
#pragma unroll
  for (int ks = 0; ks < 2; ++ks)
#pragma unroll
    for (int nt = 0; nt < 4; ++nt)
      B2[ks][nt] = load_b_frag(W2, 64, nt * 16, ks * 32, lane);
#pragma unroll
  for (int nt = 0; nt < 4; ++nt) { bias1[nt] = b1[nt * 16 + ln]; bias2[nt] = b2[nt * 16 + ln]; }

  for (int tile = wtid; tile < ntiles; tile += wstride) {
    const int m0 = tile * 16;
    prefetch_tile(X, 128, tile + wstride, ntiles, lane);
    __builtin_amdgcn_wave_barrier();
    v8f acc[4] = {};
#pragma unroll
    for (int ks = 0; ks < 4; ++ks) {
      v16bf a = load_a_frag_f32(X, 128, m0, ks * 32, lane);
#pragma unroll
      for (int nt = 0; nt < 4; ++nt)
        acc[nt] = wmma_bf16(a, load_b_frag(W1, 128, nt * 16, ks * 32, lane), acc[nt]);
    }
#pragma unroll
    for (int nt = 0; nt < 4; ++nt)
#pragma unroll
      for (int r = 0; r < 8; ++r)
        t[r + 8 * half][nt * 16 + ln] = silu_f(acc[nt][r] + bias1[nt]);
    __builtin_amdgcn_wave_barrier();
    v8f acc2[4] = {};
#pragma unroll
    for (int ks = 0; ks < 2; ++ks) {
      v16bf a = load_a_frag_f32(&t[0][0], 68, 0, ks * 32, lane);
#pragma unroll
      for (int nt = 0; nt < 4; ++nt)
        acc2[nt] = wmma_bf16(a, B2[ks][nt], acc2[nt]);
    }
#pragma unroll
    for (int r = 0; r < 8; ++r) {
      const int e = m0 + r + 8 * half;
      const int i1 = idx1[e], i2 = idx2[e];
#pragma unroll
      for (int nt = 0; nt < 4; ++nt) {
        const float v = acc2[nt][r] + bias2[nt];
        atomicAdd(node_acc + (size_t)i1 * 64 + nt * 16 + ln, v);
        atomicAdd(node_acc + (size_t)i2 * 64 + nt * 16 + ln, v);
      }
    }
  }
}

// ---------------------------------------------------------------------------
// nv1 = h_nv @ Wnv1^T (init store), nv2 = h_nv @ Wnv2^T (store); M = N*3
// Persistent waves; both weight operands held in registers.
// ---------------------------------------------------------------------------
__global__ __launch_bounds__(256) void node_vec_kernel(
    const float* __restrict__ X,                 // [N*3,64]
    const __bf16* __restrict__ W1, const __bf16* __restrict__ W2,
    float* __restrict__ nv_int, float* __restrict__ nv2, int ntiles)
{
  const int wave = threadIdx.x >> 5, lane = threadIdx.x & 31;
  const int wtid = blockIdx.x * 8 + wave;
  const int wstride = gridDim.x * 8;
  const int half = lane >> 4, ln = lane & 15;

  v16bf B1[2][4], B2[2][4];
#pragma unroll
  for (int ks = 0; ks < 2; ++ks)
#pragma unroll
    for (int nt = 0; nt < 4; ++nt) {
      B1[ks][nt] = load_b_frag(W1, 64, nt * 16, ks * 32, lane);
      B2[ks][nt] = load_b_frag(W2, 64, nt * 16, ks * 32, lane);
    }

  for (int tile = wtid; tile < ntiles; tile += wstride) {
    const int m0 = tile * 16;
    prefetch_tile(X, 64, tile + wstride, ntiles, lane);
    v8f a1[4] = {}, a2[4] = {};
#pragma unroll
    for (int ks = 0; ks < 2; ++ks) {
      v16bf a = load_a_frag_f32(X, 64, m0, ks * 32, lane);
#pragma unroll
      for (int nt = 0; nt < 4; ++nt) {
        a1[nt] = wmma_bf16(a, B1[ks][nt], a1[nt]);
        a2[nt] = wmma_bf16(a, B2[ks][nt], a2[nt]);
      }
    }
#pragma unroll
    for (int nt = 0; nt < 4; ++nt)
#pragma unroll
      for (int r = 0; r < 8; ++r) {
        const size_t m = (size_t)(m0 + r + 8 * half);
        nv_int[m * 64 + nt * 16 + ln] = a1[nt][r];
        nv2[m * 64 + nt * 16 + ln]    = a2[nt][r];
      }
  }
}

// ---------------------------------------------------------------------------
// nt1 (used only for sum-of-squares -> atomic into nt_sumsq), nt2 stored; M=N*9
// Persistent waves; both weight operands held in registers.
// ---------------------------------------------------------------------------
__global__ __launch_bounds__(256) void node_tensor_kernel(
    const float* __restrict__ X,                 // [N*9,64]
    const __bf16* __restrict__ W1, const __bf16* __restrict__ W2,
    float* __restrict__ nt2buf, float* __restrict__ nt_sumsq, int ntiles)
{
  const int wave = threadIdx.x >> 5, lane = threadIdx.x & 31;
  const int wtid = blockIdx.x * 8 + wave;
  const int wstride = gridDim.x * 8;
  const int half = lane >> 4, ln = lane & 15;

  v16bf B1[2][4], B2[2][4];
#pragma unroll
  for (int ks = 0; ks < 2; ++ks)
#pragma unroll
    for (int nt = 0; nt < 4; ++nt) {
      B1[ks][nt] = load_b_frag(W1, 64, nt * 16, ks * 32, lane);
      B2[ks][nt] = load_b_frag(W2, 64, nt * 16, ks * 32, lane);
    }

  for (int tile = wtid; tile < ntiles; tile += wstride) {
    const int m0 = tile * 16;
    prefetch_tile(X, 64, tile + wstride, ntiles, lane);
    v8f a1[4] = {}, a2[4] = {};
#pragma unroll
    for (int ks = 0; ks < 2; ++ks) {
      v16bf a = load_a_frag_f32(X, 64, m0, ks * 32, lane);
#pragma unroll
      for (int nt = 0; nt < 4; ++nt) {
        a1[nt] = wmma_bf16(a, B1[ks][nt], a1[nt]);
        a2[nt] = wmma_bf16(a, B2[ks][nt], a2[nt]);
      }
    }
#pragma unroll
    for (int r = 0; r < 8; ++r) {
      const int m = m0 + r + 8 * half;
      const int node = m / 9;
#pragma unroll
      for (int nt = 0; nt < 4; ++nt) {
        nt2buf[(size_t)m * 64 + nt * 16 + ln] = a2[nt][r];
        const float v = a1[nt][r];
        atomicAdd(nt_sumsq + (size_t)node * 64 + nt * 16 + ln, v * v);
      }
    }
  }
}

// ---------------------------------------------------------------------------
// ev1 = h_ev @ Wev1^T, fused scatter-add into nv_int (both endpoints); M = E*3
// Persistent waves; weight operand held in registers.
// ---------------------------------------------------------------------------
__global__ __launch_bounds__(256) void edge_vec_kernel(
    const float* __restrict__ X,                 // [E*3,64]
    const int* __restrict__ idx1, const int* __restrict__ idx2,
    const __bf16* __restrict__ W,                // [64,64]
    float* __restrict__ nv_int, int ntiles)      // [N*3,64]
{
  const int wave = threadIdx.x >> 5, lane = threadIdx.x & 31;
  const int wtid = blockIdx.x * 8 + wave;
  const int wstride = gridDim.x * 8;
  const int half = lane >> 4, ln = lane & 15;

  v16bf B[2][4];
#pragma unroll
  for (int ks = 0; ks < 2; ++ks)
#pragma unroll
    for (int nt = 0; nt < 4; ++nt)
      B[ks][nt] = load_b_frag(W, 64, nt * 16, ks * 32, lane);

  for (int tile = wtid; tile < ntiles; tile += wstride) {
    const int m0 = tile * 16;
    prefetch_tile(X, 64, tile + wstride, ntiles, lane);
    v8f acc[4] = {};
#pragma unroll
    for (int ks = 0; ks < 2; ++ks) {
      v16bf a = load_a_frag_f32(X, 64, m0, ks * 32, lane);
#pragma unroll
      for (int nt = 0; nt < 4; ++nt)
        acc[nt] = wmma_bf16(a, B[ks][nt], acc[nt]);
    }
#pragma unroll
    for (int r = 0; r < 8; ++r) {
      const int row = m0 + r + 8 * half;
      const int e = row / 3;
      const int c = row - e * 3;
      const size_t d1 = (size_t)(idx1[e] * 3 + c) * 64;
      const size_t d2 = (size_t)(idx2[e] * 3 + c) * 64;
#pragma unroll
      for (int nt = 0; nt < 4; ++nt) {
        const float v = acc[nt][r];
        atomicAdd(nv_int + d1 + nt * 16 + ln, v);
        atomicAdd(nv_int + d2 + nt * 16 + ln, v);
      }
    }
  }
}

// ---------------------------------------------------------------------------
// norms + concat: h_cat = [h_ns_int | ||nv_int||_axis1 | sqrt(nt_sumsq)]
// ---------------------------------------------------------------------------
__global__ void cat_build_kernel(const float* __restrict__ h_ns_int,
                                 const float* __restrict__ nv_int,
                                 const float* __restrict__ nt_sumsq,
                                 float* __restrict__ h_cat)
{
  const int i = blockIdx.x * 256 + threadIdx.x;
  if (i >= N_NODES * 64) return;
  const int node = i >> 6, c = i & 63;
  const float v0 = nv_int[(size_t)(node * 3 + 0) * 64 + c];
  const float v1 = nv_int[(size_t)(node * 3 + 1) * 64 + c];
  const float v2 = nv_int[(size_t)(node * 3 + 2) * 64 + c];
  const size_t base = (size_t)node * 192 + c;
  h_cat[base]       = h_ns_int[i];
  h_cat[base + 64]  = sqrtf(v0 * v0 + v1 * v1 + v2 * v2);
  h_cat[base + 128] = sqrtf(nt_sumsq[i]);
}

// ---------------------------------------------------------------------------
// out = silu(h_cat @ Wc1^T + bc1) @ Wc2^T + bc2   [N,192]->[N,96]->[N,256]
// ---------------------------------------------------------------------------
__global__ __launch_bounds__(256) void cat_mlp_kernel(
    const float* __restrict__ X,                 // [N,192]
    const __bf16* __restrict__ W1, const float* __restrict__ b1,  // [96,192],[96]
    const __bf16* __restrict__ W2, const float* __restrict__ b2,  // [256,96],[256]
    float* __restrict__ outbuf, int ntiles)      // [N,256]
{
  __shared__ float lds[8][16][100];
  const int wave = threadIdx.x >> 5, lane = threadIdx.x & 31;
  const int tile = blockIdx.x * 8 + wave;
  if (tile >= ntiles) return;
  const int m0 = tile * 16;
  const int half = lane >> 4, ln = lane & 15;
  float (*t)[100] = lds[wave];

  v8f acc[6] = {};
#pragma unroll
  for (int ks = 0; ks < 6; ++ks) {
    v16bf a = load_a_frag_f32(X, 192, m0, ks * 32, lane);
#pragma unroll
    for (int nt = 0; nt < 6; ++nt)
      acc[nt] = wmma_bf16(a, load_b_frag(W1, 192, nt * 16, ks * 32, lane), acc[nt]);
  }
#pragma unroll
  for (int nt = 0; nt < 6; ++nt) {
    const int n = nt * 16 + ln;
    const float bias = b1[n];
#pragma unroll
    for (int r = 0; r < 8; ++r)
      t[r + 8 * half][n] = silu_f(acc[nt][r] + bias);
  }
  __builtin_amdgcn_wave_barrier();
  v8f o[16] = {};
#pragma unroll
  for (int ks = 0; ks < 3; ++ks) {
    v16bf a = load_a_frag_f32(&t[0][0], 100, 0, ks * 32, lane);
#pragma unroll
    for (int nt = 0; nt < 16; ++nt)
      o[nt] = wmma_bf16(a, load_b_frag(W2, 96, nt * 16, ks * 32, lane), o[nt]);
  }
#pragma unroll
  for (int nt = 0; nt < 16; ++nt) {
    const int n = nt * 16 + ln;
    const float bias = b2[n];
#pragma unroll
    for (int r = 0; r < 8; ++r)
      outbuf[(size_t)(m0 + r + 8 * half) * 256 + n] = o[nt][r] + bias;
  }
}

// ---------------------------------------------------------------------------
// final gating + residuals; d_out = [ns_out | nv_out | nt_out] flat
// ---------------------------------------------------------------------------
__global__ void output_kernel(const float* __restrict__ outbuf,   // [N,256]
                              const float* __restrict__ h_ns,
                              const float* __restrict__ h_nv,
                              const float* __restrict__ h_nt,
                              const float* __restrict__ nv2,
                              const float* __restrict__ nt2,
                              float* __restrict__ out)
{
  const int i = blockIdx.x * 256 + threadIdx.x;
  if (i < N_NODES * 576) {
    const int node = i / 576;
    const int c = (i - node * 576) & 63;
    const float g = sigmoid_f(outbuf[(size_t)node * 256 + 192 + c]);
    out[(size_t)N_NODES * 320 + i] = g * nt2[i] + h_nt[i];
  }
  if (i < N_NODES * 192) {
    const int node = i / 192;
    const int c = (i - node * 192) & 63;
    const float g = sigmoid_f(outbuf[(size_t)node * 256 + 128 + c]);
    out[(size_t)N_NODES * 128 + i] = g * nv2[i] + h_nv[i];
  }
  if (i < N_NODES * 128) {
    const int node = i >> 7, c = i & 127;
    out[i] = outbuf[(size_t)node * 256 + c] + h_ns[i];
  }
}

__global__ void cvt_bf16_kernel(const float* __restrict__ src, __bf16* __restrict__ dst, int n) {
  const int i = blockIdx.x * 256 + threadIdx.x;
  if (i < n) dst[i] = (__bf16)src[i];
}

__global__ void zero_kernel(float* __restrict__ p, int n) {
  const int i = blockIdx.x * 256 + threadIdx.x;
  if (i < n) p[i] = 0.0f;
}

extern "C" void kernel_launch(void* const* d_in, const int* in_sizes, int n_in,
                              void* d_out, int out_size, void* d_ws, size_t ws_size,
                              hipStream_t stream) {
  const float* h_ns = (const float*)d_in[0];
  const float* h_nv = (const float*)d_in[1];
  const float* h_nt = (const float*)d_in[2];
  const float* h_es = (const float*)d_in[3];
  const float* h_ev = (const float*)d_in[4];
  const int*   idx1 = (const int*)d_in[5];
  const int*   idx2 = (const int*)d_in[6];
  const float* Wns1 = (const float*)d_in[7];
  const float* bns1 = (const float*)d_in[8];
  const float* Wns2 = (const float*)d_in[9];
  const float* bns2 = (const float*)d_in[10];
  const float* Wes1 = (const float*)d_in[11];
  const float* bes1 = (const float*)d_in[12];
  const float* Wes2 = (const float*)d_in[13];
  const float* bes2 = (const float*)d_in[14];
  const float* Wnv1 = (const float*)d_in[15];
  const float* Wnv2 = (const float*)d_in[16];
  const float* Wev1 = (const float*)d_in[17];
  const float* Wnt1 = (const float*)d_in[18];
  const float* Wnt2 = (const float*)d_in[19];
  const float* Wc1  = (const float*)d_in[20];
  const float* bc1  = (const float*)d_in[21];
  const float* Wc2  = (const float*)d_in[22];
  const float* bc2  = (const float*)d_in[23];

  char* ws = (char*)d_ws;
  size_t off = 0;
  auto alloc = [&](size_t bytes) -> void* {
    off = (off + 255) & ~(size_t)255;
    void* p = ws + off;
    off += bytes;
    return p;
  };

  __bf16* Wns1b = (__bf16*)alloc(64 * 128 * 2);
  __bf16* Wns2b = (__bf16*)alloc(64 * 64 * 2);
  __bf16* Wes1b = (__bf16*)alloc(64 * 128 * 2);
  __bf16* Wes2b = (__bf16*)alloc(64 * 64 * 2);
  __bf16* Wnv1b = (__bf16*)alloc(64 * 64 * 2);
  __bf16* Wnv2b = (__bf16*)alloc(64 * 64 * 2);
  __bf16* Wev1b = (__bf16*)alloc(64 * 64 * 2);
  __bf16* Wnt1b = (__bf16*)alloc(64 * 64 * 2);
  __bf16* Wnt2b = (__bf16*)alloc(64 * 64 * 2);
  __bf16* Wc1b  = (__bf16*)alloc(96 * 192 * 2);
  __bf16* Wc2b  = (__bf16*)alloc(256 * 96 * 2);

  float* h_ns_int = (float*)alloc((size_t)N_NODES * 64 * 4);
  float* nv_int   = (float*)alloc((size_t)N_NODES * 192 * 4);
  float* nv2b     = (float*)alloc((size_t)N_NODES * 192 * 4);
  float* nt2b     = (float*)alloc((size_t)N_NODES * 576 * 4);
  float* nt_sq    = (float*)alloc((size_t)N_NODES * 64 * 4);
  float* h_cat    = (float*)alloc((size_t)N_NODES * 192 * 4);
  float* outb     = (float*)alloc((size_t)N_NODES * 256 * 4);

  auto cvt = [&](const float* s, __bf16* d, int n) {
    cvt_bf16_kernel<<<(n + 255) / 256, 256, 0, stream>>>(s, d, n);
  };
  cvt(Wns1, Wns1b, 64 * 128);
  cvt(Wns2, Wns2b, 64 * 64);
  cvt(Wes1, Wes1b, 64 * 128);
  cvt(Wes2, Wes2b, 64 * 64);
  cvt(Wnv1, Wnv1b, 64 * 64);
  cvt(Wnv2, Wnv2b, 64 * 64);
  cvt(Wev1, Wev1b, 64 * 64);
  cvt(Wnt1, Wnt1b, 64 * 64);
  cvt(Wnt2, Wnt2b, 64 * 64);
  cvt(Wc1,  Wc1b,  96 * 192);
  cvt(Wc2,  Wc2b,  256 * 96);

  zero_kernel<<<(N_NODES * 64 + 255) / 256, 256, 0, stream>>>(nt_sq, N_NODES * 64);

  auto blocks_for = [](int tiles, int cap) {
    int b = (tiles + 7) / 8;
    return b < cap ? b : cap;
  };

  // Node-side GEMMs first: they initialize the scatter accumulators by plain store.
  {
    const int tiles = N_NODES / 16;                       // 1250
    node_scalar_kernel<<<blocks_for(tiles, 640), 256, 0, stream>>>(
        h_ns, Wns1b, bns1, Wns2b, bns2, h_ns_int, tiles);
  }
  {
    const int tiles = (N_NODES * 3) / 16;                 // 3750
    node_vec_kernel<<<blocks_for(tiles, 640), 256, 0, stream>>>(
        h_nv, Wnv1b, Wnv2b, nv_int, nv2b, tiles);
  }
  {
    const int tiles = (N_NODES * 9) / 16;                 // 11250
    node_tensor_kernel<<<blocks_for(tiles, 640), 256, 0, stream>>>(
        h_nt, Wnt1b, Wnt2b, nt2b, nt_sq, tiles);
  }
  // Edge-side GEMMs with fused scatter-add.
  {
    const int tiles = N_EDGES / 16;                       // 20000
    edge_scalar_kernel<<<blocks_for(tiles, 640), 256, 0, stream>>>(
        h_es, idx1, idx2, Wes1b, bes1, Wes2b, bes2, h_ns_int, tiles);
  }
  {
    const int tiles = (N_EDGES * 3) / 16;                 // 60000
    edge_vec_kernel<<<blocks_for(tiles, 640), 256, 0, stream>>>(
        h_ev, idx1, idx2, Wev1b, nv_int, tiles);
  }

  cat_build_kernel<<<(N_NODES * 64 + 255) / 256, 256, 0, stream>>>(
      h_ns_int, nv_int, nt_sq, h_cat);
  {
    const int tiles = N_NODES / 16;                       // 1250
    cat_mlp_kernel<<<(tiles + 7) / 8, 256, 0, stream>>>(
        h_cat, Wc1b, bc1, Wc2b, bc2, outb, tiles);
  }
  output_kernel<<<(N_NODES * 576 + 255) / 256, 256, 0, stream>>>(
      outb, h_ns, h_nv, h_nt, nv2b, nt2b, (float*)d_out);
}
